// NHNConv3D_8177617732124
// MI455X (gfx1250) — matrix operand order repaired and checked by reading, not compiled, not run on hardware
//
#include <hip/hip_runtime.h>

#define NPTS  100000
#define KOFF  27
#define CH    128
#define ROWS  16
#define EPSV  1e-5f

typedef __attribute__((ext_vector_type(16))) __bf16 v16bf;
typedef __attribute__((ext_vector_type(8)))  __bf16 v8bf;
typedef __attribute__((ext_vector_type(8)))  float  v8f;

__device__ __forceinline__ v16bf join8(v8bf lo, v8bf hi) {
  v16bf r;
#pragma unroll
  for (int i = 0; i < 8; ++i) { r[i] = lo[i]; r[i + 8] = hi[i]; }
  return r;
}

// ---------------------------------------------------------------------------
// Prep: Wt[k][d][c] = bf16(W[k][c][d]);  gW[k][d] = sum_c gamma[c]*W[k][c][d];
//       bW[k][d] = sum_c beta[c]*W[k][c][d]
// ---------------------------------------------------------------------------
extern "C" __global__ void nhn_prep(const float* __restrict__ W,
                                    const float* __restrict__ gamma,
                                    const float* __restrict__ beta,
                                    void* __restrict__ Wt_v,
                                    float* __restrict__ gW,
                                    float* __restrict__ bW) {
  int i = blockIdx.x * blockDim.x + threadIdx.x;     // i = k*128 + d
  if (i >= KOFF * CH) return;
  int k = i >> 7, d = i & (CH - 1);
  const float* wp = W + (size_t)k * CH * CH + d;     // stride CH over c
  __bf16* wt = (__bf16*)Wt_v + (size_t)k * CH * CH + (size_t)d * CH;
  float ga = 0.f, ba = 0.f;
#pragma unroll 8
  for (int c = 0; c < CH; ++c) {
    float w = wp[(size_t)c * CH];
    wt[c] = (__bf16)w;
    ga += gamma[c] * w;
    ba += beta[c]  * w;
  }
  gW[i] = ga;
  bW[i] = ba;
}

// ---------------------------------------------------------------------------
// Main: per block = 16 points, 128 threads (4 wave32).
// Software-pipelined: gather(k+1) issued before the WMMA burst of k,
// double-buffered A tile in LDS -> one barrier per k.
// ---------------------------------------------------------------------------
extern "C" __global__ void __launch_bounds__(128, 1)
nhn_main(const float* __restrict__ x, const void* __restrict__ Wt_v,
         const float* __restrict__ gW, const float* __restrict__ bW,
         const float* __restrict__ gamma, const float* __restrict__ bias,
         const int* __restrict__ nidx, const unsigned char* __restrict__ nmask,
         float* __restrict__ out) {
  const __bf16* __restrict__ Wt = (const __bf16*)Wt_v;

  __shared__ int    idx_lds[ROWS * KOFF];
  __shared__ float  msk_lds[ROWS * KOFF];
  __shared__ __bf16 a_lds[2][ROWS * CH];   // double-buffered bf16 A tile
  __shared__ float  o_lds[ROWS * CH];      // fp32 conv staging

  const int tid  = threadIdx.x;
  const int base = blockIdx.x * ROWS;

  // Stage neighbor indices / masks for the 16 rows
  for (int i = tid; i < ROWS * KOFF; i += 128) {
    idx_lds[i] = nidx[(size_t)base * KOFF + i];
    msk_lds[i] = nmask[(size_t)base * KOFF + i] ? 1.0f : 0.0f;
  }

  const float g = gamma[tid];
  float s1[ROWS], s2[ROWS], gk[ROWS], bk[ROWS], cnt[ROWS];
#pragma unroll
  for (int r = 0; r < ROWS; ++r) {
    s1[r] = 0.f; s2[r] = 0.f; gk[r] = 0.f; bk[r] = 0.f; cnt[r] = 0.f;
  }

  v8f acc0 = {}; v8f acc1 = {};
  const int lane = tid & 31;
  const int wave = tid >> 5;
  const int half = lane >> 4;      // which 16-lane half of the wave
  const int nlo  = lane & 15;

  __syncthreads();   // idx/msk staged

  // ---- Prologue: gather + stats for k = 0 into buffer 0 ----
  {
    const float gv0 = gW[tid];
    const float bv0 = bW[tid];
#pragma unroll
    for (int r = 0; r < ROWS; ++r) {
      const float m = msk_lds[r * KOFF + 0];
      const float v = m * x[(size_t)idx_lds[r * KOFF + 0] * CH + tid];
      s1[r]  += v;
      s2[r]  += v * v;
      gk[r]  += m * gv0;
      bk[r]  += m * bv0;
      cnt[r] += m;
      a_lds[0][r * CH + tid] = (__bf16)(v * g);
    }
  }

  for (int k = 0; k < KOFF; ++k) {
    const int p = k & 1;
    __syncthreads();   // a_lds[p] (written last iteration / prologue) is ready

    // A tiles (16x32 bf16 each) from LDS per the WMMA A lane layout:
    // lane<16: M=lane, halves = K[cb+0..7] then K[cb+16..23]
    // lane>=16: M=lane-16, halves = K[cb+8..15] then K[cb+24..31]
    v16bf A[4];
#pragma unroll
    for (int kk = 0; kk < 4; ++kk) {
      const __bf16* ap = &a_lds[p][nlo * CH + kk * 32];
      v8bf lo = *(const v8bf*)(ap + (half ? 8 : 0));
      v8bf hi = *(const v8bf*)(ap + (half ? 24 : 16));
      A[kk] = join8(lo, hi);
    }

    // Issue the k+1 gathers now so their latency hides under the WMMA burst.
    // Mask-only accumulations (cnt/gk/bk) don't depend on the loads.
    float m_[ROWS], xv[ROWS];
    if (k + 1 < KOFF) {
      const float gvn = gW[(k + 1) * CH + tid];
      const float bvn = bW[(k + 1) * CH + tid];
      __builtin_prefetch((const void*)(Wt + (size_t)(k + 1) * CH * CH +
                                       (size_t)tid * CH), 0, 1);
#pragma unroll
      for (int r = 0; r < ROWS; ++r) {
        m_[r] = msk_lds[r * KOFF + k + 1];
        xv[r] = x[(size_t)idx_lds[r * KOFF + k + 1] * CH + tid];
        cnt[r] += m_[r];
        gk[r]  += m_[r] * gvn;
        bk[r]  += m_[r] * bvn;
      }
    }

    // Each wave covers output-channel tiles 2*wave and 2*wave+1.
    // Preload all 4 B tiles of a group, then run the 4-WMMA burst.
#pragma unroll
    for (int t = 0; t < 2; ++t) {
      const int d = (wave * 2 + t) * 16 + nlo;   // output channel (B col)
      // B layout: lanes 0-15 hold K=cb+0..7 then cb+16..23, lanes 16-31 hold
      // K=cb+8..15 then cb+24..31 (matches A); Wt is [k][d][c].
      const __bf16* bp = Wt + (size_t)k * CH * CH + (size_t)d * CH + half * 8;
      v16bf Bv[4];
#pragma unroll
      for (int kk = 0; kk < 4; ++kk) {
        v8bf blo = *(const v8bf*)(bp + kk * 32);
        v8bf bhi = *(const v8bf*)(bp + kk * 32 + 16);
        Bv[kk] = join8(blo, bhi);
      }
#pragma unroll
      for (int kk = 0; kk < 4; ++kk) {
        if (t == 0)
          acc0 = __builtin_amdgcn_wmma_f32_16x16x32_bf16(
              false, A[kk], false, Bv[kk], (short)0, acc0, false, false);
        else
          acc1 = __builtin_amdgcn_wmma_f32_16x16x32_bf16(
              false, A[kk], false, Bv[kk], (short)0, acc1, false, false);
      }
    }

    // Consume the k+1 gathers: stats + bf16 store into the other buffer.
    if (k + 1 < KOFF) {
#pragma unroll
      for (int r = 0; r < ROWS; ++r) {
        const float v = m_[r] * xv[r];
        s1[r] += v;
        s2[r] += v * v;
        a_lds[1 - p][r * CH + tid] = (__bf16)(v * g);
      }
    }
  }

  // D layout -> LDS: lane n = col within tile; VGPR j = row j + 8*half.
#pragma unroll
  for (int j = 0; j < 8; ++j) {
    o_lds[(j + 8 * half) * CH + (wave * 2 + 0) * 16 + nlo] = acc0[j];
    o_lds[(j + 8 * half) * CH + (wave * 2 + 1) * 16 + nlo] = acc1[j];
  }
  __syncthreads();

  // Normalization epilogue; thread tid owns channel tid.
  const float bi = bias[tid];
#pragma unroll
  for (int r = 0; r < ROWS; ++r) {
    const float c    = cnt[r];
    const float inv  = 1.0f / c;
    const float mean = s1[r] * inv;
    const float sqm  = s2[r] * inv;
    float var = sqm - mean * mean;
    var = var < 0.f ? 0.f : var;
    const float bstd = __builtin_sqrtf(var + EPSV);
    const float vc   = c < 2.0f ? 2.0f : c;
    const float stds = __builtin_sqrtf(vc / (vc - 1.0f)) * bstd;
    const float conv = o_lds[r * CH + tid];
    out[(size_t)(base + r) * CH + tid] =
        (conv - gk[r] * mean) / stds + bk[r] + bi;
  }
}

// ---------------------------------------------------------------------------
extern "C" void kernel_launch(void* const* d_in, const int* in_sizes, int n_in,
                              void* d_out, int out_size, void* d_ws, size_t ws_size,
                              hipStream_t stream) {
  const float*         x     = (const float*)d_in[0];
  const float*         W     = (const float*)d_in[1];
  const float*         gamma = (const float*)d_in[2];
  const float*         beta  = (const float*)d_in[3];
  const float*         bias  = (const float*)d_in[4];
  const int*           nidx  = (const int*)d_in[5];
  const unsigned char* nmask = (const unsigned char*)d_in[6];
  float*               out   = (float*)d_out;

  // Workspace layout: bf16 Wt [27*128*128], then gW/bW [27*128] fp32 each.
  char*  ws = (char*)d_ws;
  void*  Wt = (void*)ws;
  float* gW = (float*)(ws + (size_t)KOFF * CH * CH * 2);
  float* bW = gW + KOFF * CH;

  nhn_prep<<<(KOFF * CH + 255) / 256, 256, 0, stream>>>(W, gamma, beta, Wt, gW, bW);
  nhn_main<<<NPTS / ROWS, 128, 0, stream>>>(x, Wt, gW, bW, gamma, bias,
                                            nidx, nmask, out);
}
